// DualStreamDecoderLayerWithRotaryEmbedding_29008209117605
// MI455X (gfx1250) — matrix operand order repaired and checked by reading, not compiled, not run on hardware
//
#include <hip/hip_runtime.h>
#include <hip/hip_bf16.h>
#include <math.h>
#include <stdint.h>

// ---------------------------------------------------------------------------
// Types for CDNA5 WMMA (wave32): v_wmma_f32_16x16x32_bf16
// ---------------------------------------------------------------------------
typedef __attribute__((ext_vector_type(16))) __bf16 v16bf;
typedef __attribute__((ext_vector_type(8)))  __bf16 v8bf;
typedef __attribute__((ext_vector_type(8)))  float  v8f;

union FragU {
    v16bf f;
    struct { v8bf lo; v8bf hi; } p;
};

__device__ __forceinline__ v8f wmma_bf16(v16bf a, v16bf b, v8f c) {
    // D = A(16x32) * B(32x16) + C, fp32 accumulate
    return __builtin_amdgcn_wmma_f32_16x16x32_bf16(
        false, a, false, b, (short)0, c, false, false);
}

// A-fragment (16x32 bf16): lane half hs picks contiguous K run [hs*16 .. hs*16+15]
__device__ __forceinline__ v16bf ldsA(const __bf16* row, int hs) {
    FragU u;
    u.p.lo = *(const v8bf*)(row + hs * 16);
    u.p.hi = *(const v8bf*)(row + hs * 16 + 8);
    return u.f;
}
// B-fragment (32x16 bf16): lane half hs picks contiguous K run [hs*16 .. hs*16+15]
__device__ __forceinline__ v16bf ldsB(const __bf16* row, int hs) {
    const __bf16* p = row + hs * 16;
    FragU u;
    u.p.lo = *(const v8bf*)(p);
    u.p.hi = *(const v8bf*)(p + 8);
    return u.f;
}
__device__ __forceinline__ v8bf cvt8(const float* p) {
    float4 a = *(const float4*)p;
    float4 b = *(const float4*)(p + 4);
    v8bf r;
    r[0] = (__bf16)a.x; r[1] = (__bf16)a.y; r[2] = (__bf16)a.z; r[3] = (__bf16)a.w;
    r[4] = (__bf16)b.x; r[5] = (__bf16)b.y; r[6] = (__bf16)b.z; r[7] = (__bf16)b.w;
    return r;
}
// B-fragment converted from an fp32 row (LDS or global), contiguous K run
__device__ __forceinline__ v16bf ldB_f32(const float* row, int hs) {
    const float* p = row + hs * 16;
    FragU u;
    u.p.lo = cvt8(p);
    u.p.hi = cvt8(p + 8);
    return u.f;
}

// ---------------------------------------------------------------------------
// CDNA5 async global->LDS copy (ASYNCcnt-tracked), 16 bytes per lane.
// LDS address = low 32 bits of the flat shared pointer.
// ---------------------------------------------------------------------------
__device__ __forceinline__ void async_g2l_b128(void* lds_dst, const void* gsrc) {
    unsigned l = (unsigned)(uintptr_t)lds_dst;
    asm volatile("global_load_async_to_lds_b128 %0, %1, off"
                 :: "v"(l), "v"(gsrc)
                 : "memory");
}
__device__ __forceinline__ void wait_async0() {
    asm volatile("s_wait_asynccnt 0x0" ::: "memory");
}

// ---------------------------------------------------------------------------
// LayerNorm over D=1024, one block (256 threads) per row
// ---------------------------------------------------------------------------
__global__ __launch_bounds__(256) void ln_kernel(const float* __restrict__ in,
                                                 float* __restrict__ out, int D) {
    const int row = blockIdx.x, tid = threadIdx.x;
    const float* p = in + (size_t)row * D;
    float4 v = *(const float4*)(p + tid * 4);
    float s = v.x + v.y + v.z + v.w;
    float q = v.x * v.x + v.y * v.y + v.z * v.z + v.w * v.w;
#pragma unroll
    for (int o = 16; o > 0; o >>= 1) {
        s += __shfl_xor(s, o);
        q += __shfl_xor(q, o);
    }
    __shared__ float ws[8], wq[8];
    const int lane = tid & 31, wv = tid >> 5;
    if (lane == 0) { ws[wv] = s; wq[wv] = q; }
    __syncthreads();
    s = 0.f; q = 0.f;
#pragma unroll
    for (int i = 0; i < 8; ++i) { s += ws[i]; q += wq[i]; }
    const float mean = s / (float)D;
    const float var = q / (float)D - mean * mean;
    const float inv = rsqrtf(var + 1e-6f);
    float4 o4;
    o4.x = (v.x - mean) * inv; o4.y = (v.y - mean) * inv;
    o4.z = (v.z - mean) * inv; o4.w = (v.w - mean) * inv;
    *(float4*)(out + (size_t)row * D + tid * 4) = o4;
}

// ---------------------------------------------------------------------------
// Generic WMMA GEMM:  C[M,N] = epi( A[M,K] @ W[N,K]^T + bias [+ resid] )
// Block = 256 threads (8 waves), block tile 128(M) x 128(N), K-step 32,
// double-buffered LDS, wave tile 32x64 (8 WMMAs / wave / K-step).
// EPI: 0 = plain(+bias), 1 = bias+SiLU, 2 = bias+residual
// ---------------------------------------------------------------------------
__device__ __forceinline__ void ldg_tile128(const float* __restrict__ base,
                                            int ldk, int k0, int tid,
                                            float4 pf[4]) {
#pragma unroll
    for (int i = 0; i < 4; ++i) {
        int idx = tid + i * 256;
        int r = idx >> 3, cc = (idx & 7) << 2;
        pf[i] = *(const float4*)(base + (size_t)r * ldk + (k0 + cc));
    }
}
__device__ __forceinline__ void sts_tile128(__bf16 (*s)[40], int tid,
                                            const float4 pf[4]) {
#pragma unroll
    for (int i = 0; i < 4; ++i) {
        int idx = tid + i * 256;
        int r = idx >> 3, cc = (idx & 7) << 2;
        s[r][cc + 0] = (__bf16)pf[i].x;
        s[r][cc + 1] = (__bf16)pf[i].y;
        s[r][cc + 2] = (__bf16)pf[i].z;
        s[r][cc + 3] = (__bf16)pf[i].w;
    }
}

template <int EPI, bool HAS_BIAS>
__global__ __launch_bounds__(256) void gemm_wmma(
    const float* __restrict__ A, const float* __restrict__ W,
    const float* __restrict__ bias, const float* __restrict__ resid,
    float* __restrict__ C, int M, int N, int K) {
    __shared__ __align__(16) __bf16 sA[2][128][40];
    __shared__ __align__(16) __bf16 sB[2][128][40];
    const int tid = threadIdx.x;
    const int lane = tid & 31, wave = tid >> 5;
    const int bm = blockIdx.y * 128, bn = blockIdx.x * 128;
    const int wm = (wave >> 1) * 32;   // 4 wave-rows  -> 128
    const int wn = (wave & 1) * 64;    // 2 wave-cols  -> 128
    const int hs = lane >> 4, rr = lane & 15;
    const float* Ab = A + (size_t)bm * K;
    const float* Wb = W + (size_t)bn * K;
    v8f acc[2][4] = {};

    const int nk = K >> 5;
    float4 pfA[4], pfB[4];
    // prologue: tile 0 -> buffer 0
    ldg_tile128(Ab, K, 0, tid, pfA);
    ldg_tile128(Wb, K, 0, tid, pfB);
    sts_tile128(sA[0], tid, pfA);
    sts_tile128(sB[0], tid, pfB);
    __syncthreads();

    for (int kt = 0; kt < nk; ++kt) {
        const int cur = kt & 1;
        const bool more = (kt + 1) < nk;
        if (more) {                       // prefetch next K-step into registers
            ldg_tile128(Ab, K, (kt + 1) << 5, tid, pfA);
            ldg_tile128(Wb, K, (kt + 1) << 5, tid, pfB);
        }
        // compute from current buffer
        v16bf a0 = ldsA(&sA[cur][wm + rr][0], hs);
        v16bf a1 = ldsA(&sA[cur][wm + 16 + rr][0], hs);
        v16bf b0 = ldsB(&sB[cur][wn + rr][0], hs);
        v16bf b1 = ldsB(&sB[cur][wn + 16 + rr][0], hs);
        v16bf b2 = ldsB(&sB[cur][wn + 32 + rr][0], hs);
        v16bf b3 = ldsB(&sB[cur][wn + 48 + rr][0], hs);
        acc[0][0] = wmma_bf16(a0, b0, acc[0][0]);
        acc[0][1] = wmma_bf16(a0, b1, acc[0][1]);
        acc[0][2] = wmma_bf16(a0, b2, acc[0][2]);
        acc[0][3] = wmma_bf16(a0, b3, acc[0][3]);
        acc[1][0] = wmma_bf16(a1, b0, acc[1][0]);
        acc[1][1] = wmma_bf16(a1, b1, acc[1][1]);
        acc[1][2] = wmma_bf16(a1, b2, acc[1][2]);
        acc[1][3] = wmma_bf16(a1, b3, acc[1][3]);
        if (more) {                       // fill the other buffer
            sts_tile128(sA[1 - cur], tid, pfA);
            sts_tile128(sB[1 - cur], tid, pfB);
        }
        __syncthreads();
    }

    // straight-line epilogue (compile-time EPI / HAS_BIAS)
#pragma unroll
    for (int mt = 0; mt < 2; ++mt) {
#pragma unroll
        for (int nt = 0; nt < 4; ++nt) {
            const int col = bn + wn + nt * 16 + rr;
            const float bv = HAS_BIAS ? bias[col] : 0.0f;
            size_t idx = (size_t)(bm + wm + mt * 16 + hs * 8) * N + col;
#pragma unroll
            for (int r = 0; r < 8; ++r) {
                float v = acc[mt][nt][r] + bv;
                if (EPI == 1) v = v * (1.0f / (1.0f + __expf(-v)));   // SiLU
                if (EPI == 2) v += resid[idx];                        // residual
                C[idx] = v;
                idx += (size_t)N;
            }
        }
    }
}

// ---------------------------------------------------------------------------
// Fused QK split + per-head RMS norm + RoPE + head-major layout; V transposed.
// Block = 32 threads (1 wave), thread i handles rotary pair (2i, 2i+1).
// Unified token order: t<512 -> c-stream (pc_*), else x-stream (px_*).
// ---------------------------------------------------------------------------
__global__ __launch_bounds__(32) void qkv_post(
    const float* __restrict__ qkbuf, const float* __restrict__ vbuf,
    const float* __restrict__ freqs,
    const float* __restrict__ pc_qn, const float* __restrict__ pc_kn,
    const float* __restrict__ px_qn, const float* __restrict__ px_kn,
    float* __restrict__ Qo, float* __restrict__ Ko, float* __restrict__ Vto) {
    const int T = 2048, S = 512, HD = 64, D = 1024;
    const int h = blockIdx.x, t = blockIdx.y, i = threadIdx.x;
    const float* qn = (t < S) ? pc_qn : px_qn;
    const float* kn = (t < S) ? pc_kn : px_kn;
    const float* qk = qkbuf + (size_t)t * 2 * D;
    float q0 = qk[h * HD + 2 * i], q1 = qk[h * HD + 2 * i + 1];
    float k0 = qk[D + h * HD + 2 * i], k1 = qk[D + h * HD + 2 * i + 1];
    float sq = q0 * q0 + q1 * q1, sk = k0 * k0 + k1 * k1;
#pragma unroll
    for (int o = 16; o > 0; o >>= 1) {
        sq += __shfl_xor(sq, o);
        sk += __shfl_xor(sk, o);
    }
    const float rq = rsqrtf(sq / (float)HD + 1e-6f);
    const float rk = rsqrtf(sk / (float)HD + 1e-6f);
    q0 = q0 * rq * qn[2 * i]; q1 = q1 * rq * qn[2 * i + 1];
    k0 = k0 * rk * kn[2 * i]; k1 = k1 * rk * kn[2 * i + 1];
    const float co = freqs[(size_t)t * 64 + 2 * i];
    const float si = freqs[(size_t)t * 64 + 2 * i + 1];
    const float qr = q0 * co - q1 * si, qi = q0 * si + q1 * co;
    const float kr = k0 * co - k1 * si, ki = k0 * si + k1 * co;
    const size_t base = ((size_t)h * T + t) * HD + 2 * i;
    Qo[base] = qr; Qo[base + 1] = qi;
    Ko[base] = kr; Ko[base + 1] = ki;
    const float v0 = vbuf[(size_t)t * D + h * HD + 2 * i];
    const float v1 = vbuf[(size_t)t * D + h * HD + 2 * i + 1];
    Vto[((size_t)h * HD + 2 * i) * T + t] = v0;      // V stored [h][d][t]
    Vto[((size_t)h * HD + 2 * i + 1) * T + t] = v1;
}

// ---------------------------------------------------------------------------
// Flash attention, causal, Hd=64, T=2048. Grid (T/128, H), 256 threads.
// Each wave owns 16 q-rows; online softmax; QK^T and P*V via WMMA bf16.
// K and V tiles are DMA'd global->LDS with CDNA5 async copies (ASYNCcnt),
// staged once per block instead of once per wave (8x traffic reduction).
// ---------------------------------------------------------------------------
__global__ __launch_bounds__(256) void attn_wmma(
    const float* __restrict__ Q, const float* __restrict__ Kin,
    const float* __restrict__ Vt, float* __restrict__ Y) {
    const int T = 2048, HD = 64, DM = 1024;
    __shared__ __align__(16) __bf16 sQ[128][72];     // Q block, bf16
    __shared__ __align__(16) __bf16 sP[8][16][72];   // per-wave P tiles
    __shared__ __align__(16) float  sKV[64][68];     // K tile [key][d] / V tile [d][k]
    const int tid = threadIdx.x, lane = tid & 31, w = tid >> 5;
    const int hs = lane >> 4, rr = lane & 15;
    const int h = blockIdx.y;
    const int qb = blockIdx.x * 128;
    const float* Qh = Q + (size_t)h * T * HD;
    const float* Kh = Kin + (size_t)h * T * HD;
    const float* Vh = Vt + (size_t)h * HD * T;

    // stage Q block (128x64 fp32 -> bf16): 2048 float4 / 256 threads
#pragma unroll
    for (int i = 0; i < 8; ++i) {
        int idx = tid + i * 256;
        int r = idx >> 4, cc = (idx & 15) << 2;
        float4 f = *(const float4*)(Qh + (size_t)(qb + r) * HD + cc);
        sQ[r][cc + 0] = (__bf16)f.x; sQ[r][cc + 1] = (__bf16)f.y;
        sQ[r][cc + 2] = (__bf16)f.z; sQ[r][cc + 3] = (__bf16)f.w;
    }
    __syncthreads();

    float mi[8], li[8];
#pragma unroll
    for (int r = 0; r < 8; ++r) { mi[r] = -1e30f; li[r] = 0.0f; }
    v8f oacc[4] = {};

    const int ktiles = (qb + 128) / 64;
    for (int kt = 0; kt < ktiles; ++kt) {
        const int kbase = kt * 64;

        // ---- async-DMA K tile (64 keys x 64 d, fp32) into LDS ----
#pragma unroll
        for (int i = 0; i < 4; ++i) {
            int idx = tid + i * 256;                  // 1024 x 16B chunks
            int r = idx >> 4, cc = (idx & 15) << 2;
            async_g2l_b128(&sKV[r][cc], Kh + (size_t)(kbase + r) * HD + cc);
        }
        wait_async0();
        __syncthreads();

        // S(16x64) = Qtile(16x64) * K(64x64)^T : 2 d-steps x 4 key tiles
        v8f s[4] = {};
#pragma unroll
        for (int d0 = 0; d0 < 64; d0 += 32) {
            v16bf aq = ldsA(&sQ[w * 16 + rr][d0], hs);
#pragma unroll
            for (int j = 0; j < 4; ++j) {
                v16bf bk = ldB_f32(&sKV[j * 16 + rr][d0], hs);
                s[j] = wmma_bf16(aq, bk, s[j]);
            }
        }
        __syncthreads();                               // done reading K tile

        // ---- async-DMA V tile (64 d x 64 k, fp32, [d][k]) into LDS ----
#pragma unroll
        for (int i = 0; i < 4; ++i) {
            int idx = tid + i * 256;
            int r = idx >> 4, cc = (idx & 15) << 2;    // r = d row
            async_g2l_b128(&sKV[r][cc], Vh + (size_t)r * T + kbase + cc);
        }

        // scale + causal mask (overlaps with V DMA)
#pragma unroll
        for (int j = 0; j < 4; ++j) {
            const int kg = kbase + j * 16 + rr;
#pragma unroll
            for (int r = 0; r < 8; ++r) {
                const int qg = qb + w * 16 + r + hs * 8;
                float v = s[j][r] * 0.125f;            // 1/sqrt(64)
                s[j][r] = (kg <= qg) ? v : -1e30f;
            }
        }
        // online softmax update (row reductions across the 16-lane group)
#pragma unroll
        for (int r = 0; r < 8; ++r) {
            float v = fmaxf(fmaxf(s[0][r], s[1][r]), fmaxf(s[2][r], s[3][r]));
            v = fmaxf(v, __shfl_xor(v, 1)); v = fmaxf(v, __shfl_xor(v, 2));
            v = fmaxf(v, __shfl_xor(v, 4)); v = fmaxf(v, __shfl_xor(v, 8));
            const float mnew = fmaxf(mi[r], v);
            const float sc = __expf(mi[r] - mnew);
            mi[r] = mnew;
            float sum = 0.0f;
#pragma unroll
            for (int j = 0; j < 4; ++j) {
                float p = __expf(s[j][r] - mnew);
                s[j][r] = p;
                sum += p;
            }
            sum += __shfl_xor(sum, 1); sum += __shfl_xor(sum, 2);
            sum += __shfl_xor(sum, 4); sum += __shfl_xor(sum, 8);
            li[r] = li[r] * sc + sum;
#pragma unroll
            for (int nt = 0; nt < 4; ++nt) oacc[nt][r] = oacc[nt][r] * sc;
        }
        // P -> per-wave LDS tile (16x64 bf16); same-wave LDS is in-order
#pragma unroll
        for (int j = 0; j < 4; ++j)
#pragma unroll
            for (int r = 0; r < 8; ++r)
                sP[w][r + hs * 8][j * 16 + rr] = (__bf16)s[j][r];

        wait_async0();
        __syncthreads();                               // V tile visible

        // O += P(16x64) * V(64x64), V B-fragments from LDS [d][k]
#pragma unroll
        for (int kk = 0; kk < 64; kk += 32) {
            v16bf ap = ldsA(&sP[w][rr][kk], hs);
#pragma unroll
            for (int nt = 0; nt < 4; ++nt) {
                v16bf bv = ldB_f32(&sKV[nt * 16 + rr][kk], hs);
                oacc[nt] = wmma_bf16(ap, bv, oacc[nt]);
            }
        }
        __syncthreads();                               // done reading V tile
    }
    // normalize and write Y[t][h*64+d] (token-major)
#pragma unroll
    for (int r = 0; r < 8; ++r) {
        const float inv = 1.0f / li[r];
        const int tok = qb + w * 16 + r + hs * 8;
#pragma unroll
        for (int nt = 0; nt < 4; ++nt)
            Y[(size_t)tok * DM + h * HD + nt * 16 + rr] = oacc[nt][r] * inv;
    }
}

// ---------------------------------------------------------------------------
// g1 *= g3 (g1 already SiLU'd by GEMM epilogue). one float4 per thread.
// ---------------------------------------------------------------------------
__global__ __launch_bounds__(256) void mul_kernel(float* __restrict__ a,
                                                  const float* __restrict__ b,
                                                  int n4) {
    int i = blockIdx.x * blockDim.x + threadIdx.x;
    if (i < n4) {
        float4 x = ((const float4*)a)[i];
        float4 y = ((const float4*)b)[i];
        x.x *= y.x; x.y *= y.y; x.z *= y.z; x.w *= y.w;
        ((float4*)a)[i] = x;
    }
}

// ---------------------------------------------------------------------------
// Host-side GEMM dispatch (compile-time epilogue selection)
// ---------------------------------------------------------------------------
static inline void launch_gemm(int epi, bool hasb, dim3 grid, hipStream_t st,
                               const float* A, const float* W, const float* bias,
                               const float* resid, float* C, int M, int N, int K) {
    if (epi == 0 && !hasb)
        gemm_wmma<0, false><<<grid, 256, 0, st>>>(A, W, bias, resid, C, M, N, K);
    else if (epi == 0)
        gemm_wmma<0, true><<<grid, 256, 0, st>>>(A, W, bias, resid, C, M, N, K);
    else if (epi == 1)
        gemm_wmma<1, true><<<grid, 256, 0, st>>>(A, W, bias, resid, C, M, N, K);
    else
        gemm_wmma<2, true><<<grid, 256, 0, st>>>(A, W, bias, resid, C, M, N, K);
}

// ---------------------------------------------------------------------------
// Host launch
// ---------------------------------------------------------------------------
extern "C" void kernel_launch(void* const* d_in, const int* in_sizes, int n_in,
                              void* d_out, int out_size, void* d_ws, size_t ws_size,
                              hipStream_t stream) {
    (void)in_sizes; (void)n_in; (void)out_size; (void)ws_size;
    const int L = 1536, S = 512, D = 1024, FF = 4096, T = 2048;

    const float* x        = (const float*)d_in[0];
    const float* c        = (const float*)d_in[1];
    const float* freqs    = (const float*)d_in[2];
    const float* px_qk_w  = (const float*)d_in[3];
    const float* px_qn    = (const float*)d_in[4];
    const float* px_kn    = (const float*)d_in[5];
    const float* px_v_w   = (const float*)d_in[6];
    const float* px_v_b   = (const float*)d_in[7];
    const float* pc_qk_w  = (const float*)d_in[8];
    const float* pc_qn    = (const float*)d_in[9];
    const float* pc_kn    = (const float*)d_in[10];
    const float* pc_v_w   = (const float*)d_in[11];
    const float* pc_v_b   = (const float*)d_in[12];
    const float* p1_proj_w = (const float*)d_in[13];
    const float* p1_proj_b = (const float*)d_in[14];
    const float* p1_w1    = (const float*)d_in[15];
    const float* p1_b1    = (const float*)d_in[16];
    const float* p1_w3    = (const float*)d_in[17];
    const float* p1_b3    = (const float*)d_in[18];
    const float* p1_w2    = (const float*)d_in[19];
    const float* p1_b2    = (const float*)d_in[20];
    const float* p2_proj_w = (const float*)d_in[21];
    const float* p2_proj_b = (const float*)d_in[22];
    const float* p2_w1    = (const float*)d_in[23];
    const float* p2_b1    = (const float*)d_in[24];
    const float* p2_w3    = (const float*)d_in[25];
    const float* p2_b3    = (const float*)d_in[26];
    const float* p2_w2    = (const float*)d_in[27];
    const float* p2_b2    = (const float*)d_in[28];

    // workspace layout (fp32), unified token order: [0,512)=c, [512,2048)=x
    float* ws = (float*)d_ws;
    size_t o = 0;
    float* hln   = ws + o; o += (size_t)T * D;        // ln(c)|ln(x)
    float* qkbuf = ws + o; o += (size_t)T * 2 * D;    // qk projections
    float* vbuf  = ws + o; o += (size_t)T * D;        // v projections
    float* Qb    = ws + o; o += (size_t)T * D;        // [h][t][64]
    float* Kb    = ws + o; o += (size_t)T * D;        // [h][t][64]
    float* Vt    = ws + o; o += (size_t)T * D;        // [h][64][t]
    float* yb    = ws + o; o += (size_t)T * D;        // attn out, token-major
    float* x1    = ws + o; o += (size_t)T * D;        // post-proj residual state
    float* h2    = ws + o; o += (size_t)T * D;        // ln(x1)
    float* g1    = ws + o; o += (size_t)T * FF;       // silu(w1)
    float* g3    = ws + o; o += (size_t)T * FF;       // w3

    float* outx = (float*)d_out;                       // x_out [1536,1024]
    float* outc = (float*)d_out + (size_t)L * D;       // c_out [512,1024]

    // 1) LayerNorm inputs
    ln_kernel<<<S, 256, 0, stream>>>(c, hln, D);
    ln_kernel<<<L, 256, 0, stream>>>(x, hln + (size_t)S * D, D);

    // 2) QK projections (N=2048)
    launch_gemm(0, false, dim3(2 * D / 128, S / 128), stream,
                hln, pc_qk_w, nullptr, nullptr, qkbuf, S, 2 * D, D);
    launch_gemm(0, false, dim3(2 * D / 128, L / 128), stream,
                hln + (size_t)S * D, px_qk_w, nullptr, nullptr,
                qkbuf + (size_t)S * 2 * D, L, 2 * D, D);

    // 3) V projections (+bias)
    launch_gemm(0, true, dim3(D / 128, S / 128), stream,
                hln, pc_v_w, pc_v_b, nullptr, vbuf, S, D, D);
    launch_gemm(0, true, dim3(D / 128, L / 128), stream,
                hln + (size_t)S * D, px_v_w, px_v_b, nullptr,
                vbuf + (size_t)S * D, L, D, D);

    // 4) RMS norm + RoPE + head split (V transposed)
    qkv_post<<<dim3(16, T), 32, 0, stream>>>(qkbuf, vbuf, freqs,
                                             pc_qn, pc_kn, px_qn, px_kn,
                                             Qb, Kb, Vt);

    // 5) causal flash attention (async K/V tile staging)
    attn_wmma<<<dim3(T / 128, 16), 256, 0, stream>>>(Qb, Kb, Vt, yb);

    // 6) output projection + residual (c uses p2, x uses p1)
    launch_gemm(2, true, dim3(D / 128, S / 128), stream,
                yb, p2_proj_w, p2_proj_b, c, x1, S, D, D);
    launch_gemm(2, true, dim3(D / 128, L / 128), stream,
                yb + (size_t)S * D, p1_proj_w, p1_proj_b, x,
                x1 + (size_t)S * D, L, D, D);

    // 7) LayerNorm
    ln_kernel<<<T, 256, 0, stream>>>(x1, h2, D);

    // 8) W1 with SiLU epilogue
    launch_gemm(1, true, dim3(FF / 128, S / 128), stream,
                h2, p2_w1, p2_b1, nullptr, g1, S, FF, D);
    launch_gemm(1, true, dim3(FF / 128, L / 128), stream,
                h2 + (size_t)S * D, p1_w1, p1_b1, nullptr,
                g1 + (size_t)S * FF, L, FF, D);

    // 9) W3
    launch_gemm(0, true, dim3(FF / 128, S / 128), stream,
                h2, p2_w3, p2_b3, nullptr, g3, S, FF, D);
    launch_gemm(0, true, dim3(FF / 128, L / 128), stream,
                h2 + (size_t)S * D, p1_w3, p1_b3, nullptr,
                g3 + (size_t)S * FF, L, FF, D);

    // 10) g1 = silu(w1) * w3
    {
        int n4 = T * FF / 4;
        mul_kernel<<<(n4 + 255) / 256, 256, 0, stream>>>(g1, g3, n4);
    }

    // 11) W2 + residual, writing straight into the concatenated output
    launch_gemm(2, true, dim3(D / 128, S / 128), stream,
                g1, p2_w2, p2_b2, x1, outc, S, D, FF);
    launch_gemm(2, true, dim3(D / 128, L / 128), stream,
                g1 + (size_t)S * FF, p1_w2, p1_b2, x1 + (size_t)S * D,
                outx, L, D, FF);
}
